// MHA_59794534695166
// MI455X (gfx1250) — compile-verified, matrix operand order checked by reading, not run on hardware
//
#include <hip/hip_runtime.h>

// Problem constants (match reference)
#define BB   16
#define NN   64
#define LL   128
#define DIN  512
#define DD   512
#define HH   8
#define HDIM 64

typedef __attribute__((ext_vector_type(16))) __bf16 v16bf_t;
typedef __attribute__((ext_vector_type(8)))  float  v8f_t;

__device__ __forceinline__ unsigned short f2bf(float x) {
    union { float f; unsigned u; } v; v.f = x;
    unsigned r = (v.u + 0x7FFFu + ((v.u >> 16) & 1u)) >> 16;  // RNE
    return (unsigned short)r;
}

// Build a v16bf fragment from a K-contiguous row in LDS.
// Layout per CDNA5 ISA (16-bit A 16x32): lane<16 holds K {cb+0..7, cb+16..23},
// lane>=16 holds K {cb+8..15, cb+24..31}. (B assumed symmetric, N via lane&15.)
__device__ __forceinline__ v16bf_t frag_krow(const unsigned short* row, int cb, int kh) {
    union { v16bf_t v; unsigned short u[16]; } r;
    const unsigned short* p0 = row + cb + 8 * kh;
    const unsigned short* p1 = row + cb + 16 + 8 * kh;
#pragma unroll
    for (int j = 0; j < 8; ++j) { r.u[j] = p0[j]; r.u[8 + j] = p1[j]; }
    return r.v;
}

// ---------------------------------------------------------------------------
// K0: wcomb[d] = sum_j w_fc2[j]*w_fc1[j,d];  wcomb[512] = sum_j w_fc2[j]*b_fc1[j] + b_fc2
__global__ void mha_wcomb_kernel(const float* __restrict__ w_fc1, const float* __restrict__ b_fc1,
                                 const float* __restrict__ w_fc2, const float* __restrict__ b_fc2,
                                 float* __restrict__ wcomb) {
    int tid = threadIdx.x;  // 512 threads
    float acc = 0.f;
#pragma unroll
    for (int j = 0; j < 16; ++j) acc += w_fc2[j] * w_fc1[j * DD + tid];
    wcomb[tid] = acc;
    if (tid == 0) {
        float c = b_fc2[0];
#pragma unroll
        for (int j = 0; j < 16; ++j) c += w_fc2[j] * b_fc1[j];
        wcomb[DD] = c;
    }
}

// ---------------------------------------------------------------------------
// K1: Q projection. Qbuf[row, d] = query[row,:] . w_q[d,:] + b_q[d], row = b*N+n
__global__ __launch_bounds__(256) void mha_qproj_kernel(const float* __restrict__ query,
                                                        const float* __restrict__ w_q,
                                                        const float* __restrict__ b_q,
                                                        float* __restrict__ Qbuf) {
    __shared__ float srow[DIN];
    const int row = blockIdx.x;
    const int tid = threadIdx.x;
    const float* q = query + (size_t)row * DIN;
    srow[tid] = q[tid]; srow[tid + 256] = q[tid + 256];
    __syncthreads();
#pragma unroll
    for (int rep = 0; rep < 2; ++rep) {
        const int d = tid + rep * 256;
        const float4* wr = (const float4*)(w_q + (size_t)d * DIN);
        const float4* sr = (const float4*)srow;
        float acc = b_q[d];
#pragma unroll 4
        for (int c4 = 0; c4 < DIN / 4; ++c4) {
            float4 w4 = wr[c4], s4 = sr[c4];
            acc += s4.x * w4.x + s4.y * w4.y + s4.z * w4.z + s4.w * w4.w;
        }
        Qbuf[(size_t)row * DD + d] = acc;
    }
}

// ---------------------------------------------------------------------------
// K2: QK[row,h,c] = (1/sqrt(hd)) * sum_{d<64} Qbuf[row, h*64+d] * w_k[h*64+d, c]
__global__ __launch_bounds__(256) void mha_qk_kernel(const float* __restrict__ Qbuf,
                                                     const float* __restrict__ w_k,
                                                     float* __restrict__ QK) {
    __shared__ float qh[HDIM];
    const int blk = blockIdx.x;         // row*8 + h
    const int row = blk >> 3, h = blk & 7;
    const int tid = threadIdx.x;
    if (tid < HDIM) qh[tid] = Qbuf[(size_t)row * DD + h * HDIM + tid] * 0.125f;  // 1/sqrt(64)
    __syncthreads();
    float acc0 = 0.f, acc1 = 0.f;
    const float* wk = w_k + (size_t)(h * HDIM) * DIN;
#pragma unroll 4
    for (int d = 0; d < HDIM; ++d) {
        const float qv = qh[d];
        acc0 += qv * wk[(size_t)d * DIN + tid];
        acc1 += qv * wk[(size_t)d * DIN + tid + 256];
    }
    float* o = QK + ((size_t)row * HH + h) * DIN;
    o[tid] = acc0; o[tid + 256] = acc1;
}

// ---------------------------------------------------------------------------
// K3: main fused attention per (n,b). Streams the 128x512 key tile once,
// WMMA bf16 for scores (QK . keys^T) and ctx (attn . keys), softmax in between,
// then folds w_v/b_v and wcomb into a single scalar e1[b,n].
#define SM_SKEYS 0        // ushort[128][512]  = 131072 B
#define SM_SQK   131072   // ushort[16][512]   =  16384 B  (aliased by sctx)
#define SM_SCTX  131072   // float [8][512]    =  16384 B
#define SM_SSC   147456   // float [16][128]   =   8192 B
#define SM_SATT  155648   // ushort[16][128]   =   4096 B
#define SM_RED   159744   // float [256]       =   1024 B
#define SM_TOTAL 160768

__global__ __launch_bounds__(256) void mha_attn_kernel(const float* __restrict__ keys,
                                                       const int* __restrict__ key_lens,
                                                       const float* __restrict__ QK,
                                                       const float* __restrict__ w_v,
                                                       const float* __restrict__ b_v,
                                                       const float* __restrict__ wcomb,
                                                       float* __restrict__ e1) {
    __shared__ __align__(16) unsigned char smem[SM_TOTAL];
    unsigned short* skeys = (unsigned short*)(smem + SM_SKEYS);
    unsigned short* sqk   = (unsigned short*)(smem + SM_SQK);
    float*          sctx  = (float*)(smem + SM_SCTX);
    float*          ssc   = (float*)(smem + SM_SSC);
    unsigned short* satt  = (unsigned short*)(smem + SM_SATT);
    float*          red   = (float*)(smem + SM_RED);

    const int blk  = blockIdx.x;           // n*16 + b  (matches keys[N,B,...] tiling)
    const int n    = blk >> 4;
    const int b    = blk & 15;
    const int tid  = threadIdx.x;
    const int wave = tid >> 5;
    const int lane = tid & 31;
    const int kh   = lane >> 4;            // lane half selects K sub-range
    const int mn   = lane & 15;            // M (A) / N (B) index within tile

    // --- stage keys tile (fp32 -> bf16) into LDS: the only HBM-heavy read ---
    const float4* kp = (const float4*)(keys + (size_t)blk * (LL * DIN));
#pragma unroll 4
    for (int i = 0; i < 64; ++i) {
        const int idx = tid + i * 256;     // float4 index, flat layout == [l][c]
        float4 v = kp[idx];
        uint2 p;
        p.x = (unsigned)f2bf(v.x) | ((unsigned)f2bf(v.y) << 16);
        p.y = (unsigned)f2bf(v.z) | ((unsigned)f2bf(v.w) << 16);
        *(uint2*)(skeys + idx * 4) = p;
    }
    // --- stage QK rows (8x512) as bf16; zero pad rows 8..15 ---
    const float* qkrow = QK + (size_t)(b * NN + n) * (HH * DIN);
#pragma unroll 4
    for (int i = 0; i < 16; ++i) {
        const int idx = tid + i * 256;
        sqk[idx] = f2bf(qkrow[idx]);
        sqk[HH * DIN + idx] = 0;
    }
    // zero attn pad rows 8..15 (flat [1024,2048))
#pragma unroll
    for (int i = 0; i < 4; ++i) satt[1024 + tid + i * 256] = 0;
    __syncthreads();

    // --- scores: wave w computes l-tile [16w,16w+16), K = 512 over 16 WMMA steps ---
    {
        const int lbase = wave * 16;
        v8f_t acc = {0.f, 0.f, 0.f, 0.f, 0.f, 0.f, 0.f, 0.f};
        const unsigned short* arow = sqk + mn * DIN;            // A: QK[m, c]
        const unsigned short* brow = skeys + (lbase + mn) * DIN; // B: keys[l, c] (K=c contiguous)
#pragma unroll
        for (int ks = 0; ks < 16; ++ks) {
            const int cb = ks * 32;
            v16bf_t a  = frag_krow(arow, cb, kh);
            v16bf_t bf = frag_krow(brow, cb, kh);
            acc = __builtin_amdgcn_wmma_f32_16x16x32_bf16(false, a, false, bf,
                                                          (short)0, acc, false, false);
        }
#pragma unroll
        for (int r = 0; r < 8; ++r)
            ssc[(r + 8 * kh) * LL + lbase + mn] = acc[r];
    }
    __syncthreads();

    // --- masked softmax: wave w == head h; b_k drops out by shift invariance ---
    {
        const int h = wave;
        const int klen = key_lens[n];
        float sv[4];
#pragma unroll
        for (int j = 0; j < 4; ++j) {
            const int l = lane + 32 * j;
            sv[j] = (l < klen) ? ssc[h * LL + l] : -1e30f;
        }
        float mx = fmaxf(fmaxf(sv[0], sv[1]), fmaxf(sv[2], sv[3]));
#pragma unroll
        for (int off = 16; off > 0; off >>= 1) mx = fmaxf(mx, __shfl_xor(mx, off, 32));
        float ev[4], sum = 0.f;
#pragma unroll
        for (int j = 0; j < 4; ++j) { ev[j] = __expf(sv[j] - mx); sum += ev[j]; }
#pragma unroll
        for (int off = 16; off > 0; off >>= 1) sum += __shfl_xor(sum, off, 32);
        const float inv = 1.f / sum;
#pragma unroll
        for (int j = 0; j < 4; ++j) satt[h * LL + lane + 32 * j] = f2bf(ev[j] * inv);
    }
    __syncthreads();

    // --- ctx: wave w owns c in [64w,64w+64): ctx[h,c] = sum_l attn[h,l]*keys[l,c] ---
#pragma unroll
    for (int nt = 0; nt < 4; ++nt) {
        const int cb2 = wave * 64 + nt * 16;
        v8f_t acc = {0.f, 0.f, 0.f, 0.f, 0.f, 0.f, 0.f, 0.f};
        const unsigned short* arow = satt + mn * LL;  // A: attn[m, l] (K=l contiguous)
#pragma unroll
        for (int kk = 0; kk < 4; ++kk) {
            const int lb = kk * 32;
            v16bf_t a = frag_krow(arow, lb, kh);
            union { v16bf_t v; unsigned short u[16]; } bb;  // B: keys[l, c] strided over l
#pragma unroll
            for (int j = 0; j < 8; ++j) {
                bb.u[j]     = skeys[(lb + 8 * kh + j) * DIN + cb2 + mn];
                bb.u[8 + j] = skeys[(lb + 16 + 8 * kh + j) * DIN + cb2 + mn];
            }
            acc = __builtin_amdgcn_wmma_f32_16x16x32_bf16(false, a, false, bb.v,
                                                          (short)0, acc, false, false);
        }
#pragma unroll
        for (int r = 0; r < 8; ++r) {
            const int m = r + 8 * kh;
            if (m < HH) sctx[m * DIN + cb2 + mn] = acc[r];  // aliases sqk (dead now)
        }
    }
    __syncthreads();

    // --- out[dd] = ctx[h(dd),:].w_v[dd,:] + b_v[dd]; fold into e1 = out . wcomb ---
    float part = 0.f;
#pragma unroll
    for (int rep = 0; rep < 2; ++rep) {
        const int dd = tid + rep * 256;
        const int h = dd >> 6;
        const float4* wv = (const float4*)(w_v + (size_t)dd * DIN);
        const float4* cx = (const float4*)(sctx + h * DIN);
        float acc = b_v[dd];
#pragma unroll 4
        for (int c4 = 0; c4 < DIN / 4; ++c4) {
            float4 w4 = wv[c4], x4 = cx[c4];
            acc += w4.x * x4.x + w4.y * x4.y + w4.z * x4.z + w4.w * x4.w;
        }
        part += acc * wcomb[dd];
    }
    red[tid] = part;
    __syncthreads();
    for (int s = 128; s > 0; s >>= 1) {
        if (tid < s) red[tid] += red[tid + s];
        __syncthreads();
    }
    if (tid == 0) e1[b * NN + n] = red[0];
}

// ---------------------------------------------------------------------------
// K4: out[b,m] = sum_n intensity[b,n,m] * e1[b,n] + cconst
__global__ __launch_bounds__(256) void mha_final_kernel(const float* __restrict__ intensity,
                                                        const float* __restrict__ e1,
                                                        const float* __restrict__ wcomb,
                                                        float* __restrict__ out) {
    const int i = blockIdx.x * 256 + threadIdx.x;   // [0,1024)
    const int b = i >> 6, m = i & 63;
    float acc = wcomb[DD];                          // cconst
    const float* it = intensity + (size_t)b * (NN * NN);
    const float* eb = e1 + b * NN;
#pragma unroll 4
    for (int nn = 0; nn < NN; ++nn) acc += it[nn * NN + m] * eb[nn];
    out[i] = acc;
}

// ---------------------------------------------------------------------------
extern "C" void kernel_launch(void* const* d_in, const int* in_sizes, int n_in,
                              void* d_out, int out_size, void* d_ws, size_t ws_size,
                              hipStream_t stream) {
    (void)in_sizes; (void)n_in; (void)out_size; (void)ws_size;
    const float* query     = (const float*)d_in[0];
    const float* keys      = (const float*)d_in[1];
    const int*   key_lens  = (const int*)d_in[2];
    const float* intensity = (const float*)d_in[3];
    const float* w_q       = (const float*)d_in[4];
    const float* b_q       = (const float*)d_in[5];
    const float* w_k       = (const float*)d_in[6];
    // d_in[7] = b_k: provably drops out of softmax (per-head shift) -> unused
    const float* w_v       = (const float*)d_in[8];
    const float* b_v       = (const float*)d_in[9];
    const float* w_fc1     = (const float*)d_in[10];
    const float* b_fc1     = (const float*)d_in[11];
    const float* w_fc2     = (const float*)d_in[12];
    const float* b_fc2     = (const float*)d_in[13];

    float* ws    = (float*)d_ws;
    float* wcomb = ws;                               // 513 floats (wcomb + cconst)
    float* Qbuf  = ws + 1024;                        // 1024*512
    float* QKbuf = Qbuf + (size_t)BB * NN * DD;      // 1024*8*512
    float* e1    = QKbuf + (size_t)BB * NN * HH * DIN; // 1024

    mha_wcomb_kernel<<<1, 512, 0, stream>>>(w_fc1, b_fc1, w_fc2, b_fc2, wcomb);
    mha_qproj_kernel<<<BB * NN, 256, 0, stream>>>(query, w_q, b_q, Qbuf);
    mha_qk_kernel<<<BB * NN * HH, 256, 0, stream>>>(Qbuf, w_k, QKbuf);
    mha_attn_kernel<<<NN * BB, 256, 0, stream>>>(keys, key_lens, QKbuf, w_v, b_v, wcomb, e1);
    mha_final_kernel<<<4, 256, 0, stream>>>(intensity, e1, wcomb, (float*)d_out);
}
